// MolCLREncoder_66580583022640
// MI455X (gfx1250) — matrix operand order repaired and checked by reading, not compiled, not run on hardware
//
#include <hip/hip_runtime.h>

// ---------------------------------------------------------------------------
// Types for CDNA5 WMMA (wave32): v_wmma_f32_16x16x32_bf16
// ---------------------------------------------------------------------------
typedef __attribute__((ext_vector_type(16))) __bf16       v16bf;
typedef __attribute__((ext_vector_type(8)))  float        v8f;
typedef __attribute__((ext_vector_type(4)))  unsigned int u32x4;
typedef __attribute__((ext_vector_type(4)))  float        f32x4;

union BFrag { v16bf v; u32x4 u[2]; };

#define BN_EPS 1e-5f

__device__ __forceinline__ v8f wmma_bf16(v16bf a, v16bf b, v8f c) {
    // (neg_a, A, neg_b, B, c_mod, C, reuse_a, reuse_b)
    return __builtin_amdgcn_wmma_f32_16x16x32_bf16(false, a, false, b,
                                                   (short)0, c, false, false);
}

// ---------------------------------------------------------------------------
// Utility kernels
// ---------------------------------------------------------------------------
__global__ void k_zero(float* __restrict__ p, long long n) {
    long long i = (long long)blockIdx.x * blockDim.x + threadIdx.x;
    if (i < n) p[i] = 0.f;
}

__global__ void k_deg(const int* __restrict__ dst, float* __restrict__ deg, int E) {
    int e = blockIdx.x * blockDim.x + threadIdx.x;
    if (e < E) atomicAdd(&deg[dst[e]], 1.0f);
}

__global__ void k_dis(float* __restrict__ deg, int N) {
    int i = blockIdx.x * blockDim.x + threadIdx.x;
    if (i < N) deg[i] = rsqrtf(deg[i] + 1.0f);  // +1 self-loop; deg >= 1 always
}

__global__ void k_norm(const int* __restrict__ src, const int* __restrict__ dst,
                       const float* __restrict__ dis, float* __restrict__ nrm, int E) {
    int e = blockIdx.x * blockDim.x + threadIdx.x;
    if (e < E) nrm[e] = dis[src[e]] * dis[dst[e]];
}

// f32 (M x K) -> bf16 (M x Kp), zero-padding columns K..Kp-1
__global__ void k_cvt_bf16(const float* __restrict__ X, __bf16* __restrict__ Y,
                           long long total, int K, int Kp) {
    long long t = (long long)blockIdx.x * blockDim.x + threadIdx.x;
    if (t >= total) return;
    int c = (int)(t % Kp);
    long long r = t / Kp;
    float v = (c < K) ? X[r * K + c] : 0.f;
    Y[t] = (__bf16)v;
}

// Repack weight W (K x Nout, row-major f32) into per-lane-contiguous bf16 B
// fragments for v_wmma_f32_16x16x32_bf16:
//   frag[(kt*nT + nt)*512 + lane*16 + j] = W[kt*32 + (lane/16)*16 + j][nt*16 + lane%16]
__global__ void k_repack(const float* __restrict__ W, __bf16* __restrict__ Wp,
                         int K, int Kp, int Nout) {
    int idx = blockIdx.x * blockDim.x + threadIdx.x;
    int total = Kp * Nout;
    if (idx >= total) return;
    int j    = idx & 15;
    int lane = (idx >> 4) & 31;
    int tile = idx >> 9;
    int nT = Nout >> 4;
    int kt = tile / nT, nt = tile - kt * nT;
    int k = kt * 32 + ((lane >> 4) << 4) + j;
    int n = (nt << 4) + (lane & 15);
    float v = (k < K) ? W[(long long)k * Nout + n] : 0.f;
    Wp[idx] = (__bf16)v;
}

// ---------------------------------------------------------------------------
// WMMA GEMM with async LDS weight staging:
//   C[M x Nout] = A[M x Kp](bf16 row-major) * Wrepacked (+ bias)
// Block = 256 threads (8 waves). Block owns a 64-column slab; B fragments for
// all k-tiles are staged global->LDS once via GLOBAL_LOAD_ASYNC_TO_LDS_B128,
// then each wave computes a 16(M) x 64(N) tile reading B via ds_load_b128.
// grid = (ceil(M/16 / 8), Nout/64), block = 256. Kp/32 <= 8 (<=32KB LDS).
// ---------------------------------------------------------------------------
__global__ __launch_bounds__(256)
void k_gemm(const __bf16* __restrict__ A, const __bf16* __restrict__ Wp,
            const float* __restrict__ bias, float* __restrict__ C,
            int M, int Kp, int Nout) {
    __shared__ u32x4 smem[2048];            // 32 KB: [kt][tile(4)][lane(32)][2]

    int tid  = threadIdx.x;
    int lane = tid & 31;
    int w    = tid >> 5;                     // wave id 0..7
    int n0   = blockIdx.y << 6;
    int nT   = Nout >> 4;
    int kT   = Kp >> 5;

    // ---- stage this n-slab's repacked weights into LDS (async DMA) --------
    {
        unsigned smbase = (unsigned)(unsigned long long)&smem[0];
        const char* wblk = (const char*)(Wp + (long long)(n0 >> 4) * 512);
        for (int kt = 0; kt < kT; ++kt) {
            unsigned ldsoff = smbase + (unsigned)(kt * 4096 + tid * 16);
            const void* g = wblk + (long long)kt * nT * 1024 + tid * 16;
            asm volatile("global_load_async_to_lds_b128 %0, %1, off"
                         :: "v"(ldsoff), "v"(g) : "memory");
        }
        asm volatile("s_wait_asynccnt 0x0" ::: "memory");
    }
    __syncthreads();

    int mtile = blockIdx.x * 8 + w;
    if (mtile * 16 < M) {                    // wave-uniform guard (ragged M)
        int h = lane >> 4;                   // half-wave selector
        int m = lane & 15;

        v8f acc0 = {}, acc1 = {}, acc2 = {}, acc3 = {};

        // A fragment: elems 0..15 = K[k0+16h .. +15]
        const __bf16* ap = A + (long long)(mtile * 16 + m) * Kp + (h << 4);
        int bidx = (lane << 1);              // [lane][2] within a tile
        for (int kt = 0; kt < kT; ++kt) {
            BFrag a;
            a.u[0] = *(const u32x4*)(ap);
            a.u[1] = *(const u32x4*)(ap + 8);
            ap += 32;
            __builtin_prefetch(ap, 0, 3);    // global_prefetch_b8, near caches

            int kb = (kt << 8) + bidx;       // kt*256 u32x4 units
            BFrag b0, b1, b2, b3;
            b0.u[0] = smem[kb];       b0.u[1] = smem[kb + 1];
            b1.u[0] = smem[kb + 64];  b1.u[1] = smem[kb + 65];
            b2.u[0] = smem[kb + 128]; b2.u[1] = smem[kb + 129];
            b3.u[0] = smem[kb + 192]; b3.u[1] = smem[kb + 193];

            acc0 = wmma_bf16(a.v, b0.v, acc0);
            acc1 = wmma_bf16(a.v, b1.v, acc1);
            acc2 = wmma_bf16(a.v, b2.v, acc2);
            acc3 = wmma_bf16(a.v, b3.v, acc3);
        }

        int cn = n0 + m;
        float bv0 = bias ? bias[cn]      : 0.f;
        float bv1 = bias ? bias[cn + 16] : 0.f;
        float bv2 = bias ? bias[cn + 32] : 0.f;
        float bv3 = bias ? bias[cn + 48] : 0.f;
        // C/D layout: VGPR r -> row 16*mtile + r + 8*h, col n0 + lane%16
        for (int r = 0; r < 8; ++r) {
            long long row = mtile * 16 + r + (h << 3);
            float* out = C + row * Nout + cn;
            out[0]  = acc0[r] + bv0;
            out[16] = acc1[r] + bv1;
            out[32] = acc2[r] + bv2;
            out[48] = acc3[r] + bv3;
        }
    }
}

// ---------------------------------------------------------------------------
// GCN aggregation
// ---------------------------------------------------------------------------
// OUT[i][c] = dis[i]^2 * H[i][c] + b[c]    (self-loop term; also zero-inits OUT)
__global__ void k_self_init(const float* __restrict__ H, const float* __restrict__ dis,
                            const float* __restrict__ b, float* __restrict__ OUT,
                            long long total, int C) {
    long long t = (long long)blockIdx.x * blockDim.x + threadIdx.x;
    if (t >= total) return;
    int c = (int)(t % C);
    long long i = t / C;
    float d = dis[i];
    OUT[t] = d * d * H[t] + b[c];
}

// OUT[dst] += norm[e] * H[src]   (float4 chunks, f32 global atomics)
__global__ void k_scatter(const int* __restrict__ src, const int* __restrict__ dst,
                          const float* __restrict__ nrm, const float* __restrict__ H,
                          float* __restrict__ OUT, long long total, int C) {
    long long t = (long long)blockIdx.x * blockDim.x + threadIdx.x;
    if (t >= total) return;
    int Cq = C >> 2;
    int q = (int)(t % Cq);
    long long e = t / Cq;
    int s = src[e], d = dst[e];
    float w = nrm[e];
    f32x4 hv = *(const f32x4*)(H + (long long)s * C + (q << 2));
    float* o = OUT + (long long)d * C + (q << 2);
    atomicAdd(o + 0, w * hv[0]);
    atomicAdd(o + 1, w * hv[1]);
    atomicAdd(o + 2, w * hv[2]);
    atomicAdd(o + 3, w * hv[3]);
}

// ---------------------------------------------------------------------------
// BatchNorm (training mode, biased variance) + ReLU
// ---------------------------------------------------------------------------
// blockDim.x == C ; each block owns a row slab, thread t owns column t (coalesced)
__global__ void k_bn_partial(const float* __restrict__ X, float* __restrict__ sacc,
                             float* __restrict__ qacc, int Nrows, int C, int rpb) {
    int c = threadIdx.x;
    long long r0 = (long long)blockIdx.x * rpb;
    long long r1 = r0 + rpb; if (r1 > Nrows) r1 = Nrows;
    float s = 0.f, q = 0.f;
    for (long long r = r0; r < r1; ++r) {
        float v = X[r * C + c];
        s += v; q += v * v;
    }
    atomicAdd(&sacc[c], s);
    atomicAdd(&qacc[c], q);
}

__global__ void k_bn_final(float* __restrict__ sacc, float* __restrict__ qacc,
                           const float* __restrict__ g, const float* __restrict__ be,
                           int C, float invN) {
    int c = blockIdx.x * blockDim.x + threadIdx.x;
    if (c >= C) return;
    float m = sacc[c] * invN;
    float v = qacc[c] * invN - m * m;
    float sc = g[c] * rsqrtf(v + BN_EPS);
    sacc[c] = sc;                 // scale
    qacc[c] = be[c] - m * sc;     // shift
}

__global__ void k_bn_apply_relu(float* __restrict__ X, const float* __restrict__ sc,
                                const float* __restrict__ sh, long long total, int C) {
    long long t = (long long)blockIdx.x * blockDim.x + threadIdx.x;
    if (t >= total) return;
    int c = (int)(t % C);
    float y = X[t] * sc[c] + sh[c];
    X[t] = y > 0.f ? y : 0.f;
}

// ---------------------------------------------------------------------------
// Global mean pool over sorted graph ids
// ---------------------------------------------------------------------------
__global__ void k_pool_acc(const float* __restrict__ H, const int* __restrict__ batch,
                           float* __restrict__ sums, long long total, int C) {
    long long t = (long long)blockIdx.x * blockDim.x + threadIdx.x;
    if (t >= total) return;
    int Cq = C >> 2;
    int q = (int)(t % Cq);
    long long i = t / Cq;
    int g = batch[i];
    f32x4 hv = *(const f32x4*)(H + i * C + (q << 2));
    float* o = sums + (long long)g * C + (q << 2);
    atomicAdd(o + 0, hv[0]);
    atomicAdd(o + 1, hv[1]);
    atomicAdd(o + 2, hv[2]);
    atomicAdd(o + 3, hv[3]);
}

__global__ void k_pool_cnt(const int* __restrict__ batch, float* __restrict__ cnt, int N) {
    int i = blockIdx.x * blockDim.x + threadIdx.x;
    if (i < N) atomicAdd(&cnt[batch[i]], 1.0f);
}

__global__ void k_pool_div_bf16(const float* __restrict__ sums, const float* __restrict__ cnt,
                                __bf16* __restrict__ out, long long total, int C) {
    long long t = (long long)blockIdx.x * blockDim.x + threadIdx.x;
    if (t >= total) return;
    long long g = t / C;
    float d = cnt[g];
    d = d > 1.f ? d : 1.f;
    out[t] = (__bf16)(sums[t] / d);
}

// ---------------------------------------------------------------------------
// Host-side orchestration
// ---------------------------------------------------------------------------
static inline unsigned nblk(long long n, int b) { return (unsigned)((n + b - 1) / b); }

extern "C" void kernel_launch(void* const* d_in, const int* in_sizes, int n_in,
                              void* d_out, int out_size, void* d_ws, size_t ws_size,
                              hipStream_t stream) {
    // Inputs (setup_inputs order)
    const float* x     = (const float*)d_in[0];
    const int*   ei    = (const int*)d_in[1];
    const int*   batch = (const int*)d_in[2];
    const float* W1 = (const float*)d_in[4];
    const float* b1 = (const float*)d_in[5];
    const float* g1 = (const float*)d_in[6];
    const float* e1 = (const float*)d_in[7];
    const float* W2 = (const float*)d_in[8];
    const float* b2 = (const float*)d_in[9];
    const float* g2 = (const float*)d_in[10];
    const float* e2 = (const float*)d_in[11];
    const float* W3 = (const float*)d_in[12];
    const float* b3 = (const float*)d_in[13];
    const float* g3 = (const float*)d_in[14];
    const float* e3 = (const float*)d_in[15];
    const float* Wp = (const float*)d_in[16];
    const float* bp = (const float*)d_in[17];

    const int N   = in_sizes[2];               // nodes
    const int E   = in_sizes[1] / 2;           // edges (without self loops)
    const int H1  = in_sizes[5];               // 128
    const int H2  = in_sizes[9];               // 256
    const int P   = in_sizes[17];              // 256
    const int Fin = in_sizes[4] / H1;          // 9
    const int G   = out_size / P;              // 8192
    const int Kp1 = ((Fin + 31) / 32) * 32;    // 32

    const int* src = ei;
    const int* dst = ei + E;

    // Workspace carve-out
    char* wsp = (char*)d_ws;
    size_t off = 0;
    auto carve = [&](size_t bytes) -> void* {
        void* p = wsp + off;
        off = (off + bytes + 255) & ~(size_t)255;
        return p;
    };
    float*  bufA   = (float*)carve((size_t)N * H2 * 4);      // GEMM output / gather src
    float*  bufB   = (float*)carve((size_t)N * H2 * 4);      // scatter dest / activation
    __bf16* actbf  = (__bf16*)carve((size_t)N * H2 * 2);     // bf16 activations
    float*  dis    = (float*)carve((size_t)N * 4);           // degree -> rsqrt(deg)
    float*  normE  = (float*)carve((size_t)E * 4);
    __bf16* W1p    = (__bf16*)carve((size_t)Kp1 * H1 * 2);
    __bf16* W2p    = (__bf16*)carve((size_t)H1 * H2 * 2);
    __bf16* W3p    = (__bf16*)carve((size_t)H2 * H2 * 2);
    __bf16* Wpp    = (__bf16*)carve((size_t)H2 * P * 2);
    float*  stats  = (float*)carve(512 * 4);                 // sum/scale, sq/shift
    float*  sums   = (float*)carve(((size_t)G * P + G) * 4); // pooled sums + counts
    float*  cnts   = sums + (size_t)G * P;
    __bf16* poolbf = (__bf16*)carve((size_t)G * P * 2);
    (void)ws_size; (void)n_in;

    float* ssum = stats;
    float* ssq  = stats + 256;

    // --- degree / norm -----------------------------------------------------
    k_zero<<<nblk(N, 256), 256, 0, stream>>>(dis, N);
    k_deg<<<nblk(E, 256), 256, 0, stream>>>(dst, dis, E);
    k_dis<<<nblk(N, 256), 256, 0, stream>>>(dis, N);
    k_norm<<<nblk(E, 256), 256, 0, stream>>>(src, dst, dis, normE, E);

    // --- repack weights to WMMA B-fragment layout --------------------------
    k_repack<<<nblk((long long)Kp1 * H1, 256), 256, 0, stream>>>(W1, W1p, Fin, Kp1, H1);
    k_repack<<<nblk((long long)H1 * H2, 256), 256, 0, stream>>>(W2, W2p, H1, H1, H2);
    k_repack<<<nblk((long long)H2 * H2, 256), 256, 0, stream>>>(W3, W3p, H2, H2, H2);
    k_repack<<<nblk((long long)H2 * P, 256), 256, 0, stream>>>(Wp, Wpp, H2, H2, P);

    // --- one GCN layer: act(f32,Cin) -> bufB(f32,Cout) ---------------------
    auto layer = [&](const float* act, int Cin, int Kp, const __bf16* Wrep, int Cout,
                     const float* b, const float* gm, const float* bt) {
        long long tcvt = (long long)N * Kp;
        k_cvt_bf16<<<nblk(tcvt, 256), 256, 0, stream>>>(act, actbf, tcvt, Cin, Kp);
        dim3 gg((unsigned)((N / 16 + 7) / 8), (unsigned)(Cout / 64));
        k_gemm<<<gg, 256, 0, stream>>>(actbf, Wrep, nullptr, bufA, N, Kp, Cout);
        long long tn = (long long)N * Cout;
        k_self_init<<<nblk(tn, 256), 256, 0, stream>>>(bufA, dis, b, bufB, tn, Cout);
        long long te = (long long)E * (Cout >> 2);
        k_scatter<<<nblk(te, 256), 256, 0, stream>>>(src, dst, normE, bufA, bufB, te, Cout);
        // BatchNorm + ReLU
        k_zero<<<2, 256, 0, stream>>>(stats, 512);
        int rpb = (N + 511) / 512;
        k_bn_partial<<<512, Cout, 0, stream>>>(bufB, ssum, ssq, N, Cout, rpb);
        k_bn_final<<<1, 256, 0, stream>>>(ssum, ssq, gm, bt, Cout, 1.0f / (float)N);
        k_bn_apply_relu<<<nblk(tn, 256), 256, 0, stream>>>(bufB, ssum, ssq, tn, Cout);
    };

    layer(x,    Fin, Kp1, W1p, H1, b1, g1, e1);   // -> bufB [N x 128]
    layer(bufB, H1,  H1,  W2p, H2, b2, g2, e2);   // -> bufB [N x 256]
    layer(bufB, H2,  H2,  W3p, H2, b3, g3, e3);   // -> bufB [N x 256]

    // --- global mean pool --------------------------------------------------
    long long tpool = (long long)G * P + G;
    k_zero<<<nblk(tpool, 256), 256, 0, stream>>>(sums, tpool);
    long long tacc = (long long)N * (P >> 2);
    k_pool_acc<<<nblk(tacc, 256), 256, 0, stream>>>(bufB, batch, sums, tacc, P);
    k_pool_cnt<<<nblk(N, 256), 256, 0, stream>>>(batch, cnts, N);
    long long tgp = (long long)G * P;
    k_pool_div_bf16<<<nblk(tgp, 256), 256, 0, stream>>>(sums, cnts, poolbf, tgp, P);

    // --- final projection: pooled @ Wp + bp -> d_out (f32) ------------------
    dim3 gg((unsigned)((G / 16 + 7) / 8), (unsigned)(P / 64));
    k_gemm<<<gg, 256, 0, stream>>>(poolbf, Wpp, bp, (float*)d_out, G, H2, P);
}
